// RouteGNN_42245298323840
// MI455X (gfx1250) — compile-verified, matrix operand order checked by reading, not run on hardware
//
#include <hip/hip_runtime.h>
#include <hip/hip_bf16.h>

typedef __attribute__((ext_vector_type(16))) __bf16 v16bf;
typedef __attribute__((ext_vector_type(8)))  float  v8f;
typedef __attribute__((ext_vector_type(4)))  int    v4i;

// ---------------------------------------------------------------------------
// Native f32 global atomic add (no-return -> STOREcnt, no CAS loop fallback)
// ---------------------------------------------------------------------------
__device__ __forceinline__ void gatomic_fadd(float* p, float v) {
    asm volatile("global_atomic_add_f32 %0, %1, off"
                 :
                 : "v"((unsigned long long)p), "v"(v)
                 : "memory");
}

// ---------------------------------------------------------------------------
// Encoder: O[r][n] = sum_k X[r][k] * W[k][n] + b[n]   (K <= 34, H = 128)
// ---------------------------------------------------------------------------
__global__ __launch_bounds__(128)
void encode_kernel(const float* __restrict__ X, const float* __restrict__ W,
                   const float* __restrict__ b, float* __restrict__ O,
                   int nRows, int K) {
    __shared__ float Wl[34 * 128];
    const int t = threadIdx.x;
    for (int i = t; i < K * 128; i += 128) Wl[i] = W[i];
    __syncthreads();
    const float bv = b[t];
    int r0 = blockIdx.x * 32;
    int r1 = r0 + 32; if (r1 > nRows) r1 = nRows;
    for (int r = r0; r < r1; ++r) {
        const float* x = X + (size_t)r * K;
        float acc = bv;
        for (int k = 0; k < K; ++k) acc += x[k] * Wl[k * 128 + t];
        O[(size_t)r * 128 + t] = acc;
    }
}

// ---------------------------------------------------------------------------
// Fill with value
// ---------------------------------------------------------------------------
__global__ void fillv_kernel(float* __restrict__ p, float v, size_t n) {
    size_t i = (size_t)blockIdx.x * blockDim.x + threadIdx.x;
    if (i < n) p[i] = v;
}

// ---------------------------------------------------------------------------
// Degree count + reciprocal (1/max(cnt,1))
// ---------------------------------------------------------------------------
__global__ void count_kernel(const int* __restrict__ idx, float* __restrict__ cnt, int E) {
    int e = blockIdx.x * blockDim.x + threadIdx.x;
    if (e < E) gatomic_fadd(cnt + idx[e], 1.0f);
}

__global__ void recip_kernel(float* __restrict__ c, int n) {
    int i = blockIdx.x * blockDim.x + threadIdx.x;
    if (i < n) c[i] = 1.0f / fmaxf(c[i], 1.0f);
}

// ---------------------------------------------------------------------------
// Edge scatter-add: agg[dst[e]][:] += X[src[e]][:]
// One wave (32 lanes) per edge; float4 per lane; 4 native f32 atomics.
// ---------------------------------------------------------------------------
__global__ __launch_bounds__(256)
void scatter_add_kernel(const float* __restrict__ X, const int* __restrict__ src,
                        const int* __restrict__ dst, float* __restrict__ agg, int E) {
    long long t = (long long)blockIdx.x * blockDim.x + threadIdx.x;
    int e = (int)(t >> 5);
    if (e >= E) return;
    int c = (int)(t & 31) * 4;
    const float4 v = *(const float4*)(X + (size_t)src[e] * 128 + c);
    float* p = agg + (size_t)dst[e] * 128 + c;
    gatomic_fadd(p + 0, v.x);
    gatomic_fadd(p + 1, v.y);
    gatomic_fadd(p + 2, v.z);
    gatomic_fadd(p + 3, v.w);
}

// ---------------------------------------------------------------------------
// Pack a 128x128 f32 weight (row-major, W[k][n]) into WMMA B-fragment layout
// (bf16): element s of lane's fragment for tile (nt,kt):
//   K = kt*32 + (s<8?0:16) + (lane<16?0:8) + (s&7),  N = nt*16 + (lane&15)
// per the 16-bit 32x16 B layout of v_wmma_f32_16x16x32_bf16 (ISA 7.12.2).
// ---------------------------------------------------------------------------
__global__ __launch_bounds__(256)
void pack_w_kernel(const float* __restrict__ W, __bf16* __restrict__ P) {
    int t = blockIdx.x * blockDim.x + threadIdx.x;  // 0..1023
    if (t >= 1024) return;
    const int lane = t & 31;
    const int kt   = (t >> 5) & 3;
    const int nt   = t >> 7;
    const int col  = nt * 16 + (lane & 15);
    const int hs   = (lane < 16) ? 0 : 8;
    __bf16* o = P + (size_t)t * 16;
#pragma unroll
    for (int s = 0; s < 16; ++s) {
        int k = kt * 32 + ((s < 8) ? 0 : 16) + hs + (s & 7);
        o[s] = (__bf16)W[k * 128 + col];
    }
}

// ---------------------------------------------------------------------------
// WMMA GEMM: OUT[r][0:128] (op)= (scale[r] * A[r][:]) @ W + bias
// Block = 128 threads = 4 waves. Packed W (32KB) staged to LDS once per
// block via GLOBAL_LOAD_ASYNC_TO_LDS_B128 (ASYNCcnt) when available.
// Each wave: 16-row strip, 4 resident bf16 A-fragments, 8 N-tiles x 4
// K-steps = 32 v_wmma_f32_16x16x32_bf16, B-fragments preloaded from LDS
// before each WMMA chain so ds loads overlap matrix math.
// Row counts are multiples of 16 -> EXEC all-ones for every WMMA.
// ---------------------------------------------------------------------------
template <bool ACC, bool RELU>
__global__ __launch_bounds__(128)
void gemm_wmma_kernel(const float* __restrict__ A, const __bf16* __restrict__ Wpk,
                      const float* __restrict__ bias, const float* __restrict__ scale,
                      float* __restrict__ OUT, int nRows) {
    __shared__ __align__(64) __bf16 Wlds[128 * 128];
    const int tid = threadIdx.x;

    // ---- stage packed W into LDS (all waves participate, then barrier) ----
    {
        const v4i* g = (const v4i*)Wpk;   // 2048 x 16B
        v4i*       l = (v4i*)Wlds;
#if __has_builtin(__builtin_amdgcn_global_load_async_to_lds_b128)
        typedef __attribute__((address_space(1))) v4i* gv4i_p;
        typedef __attribute__((address_space(3))) v4i* sv4i_p;
#pragma unroll
        for (int i = 0; i < 16; ++i) {
            int idx = tid + i * 128;
            __builtin_amdgcn_global_load_async_to_lds_b128(
                (gv4i_p)(g + idx), (sv4i_p)(l + idx), 0, 0);
        }
#if __has_builtin(__builtin_amdgcn_s_wait_asynccnt)
        __builtin_amdgcn_s_wait_asynccnt(0);
#else
        asm volatile("s_wait_asynccnt 0x0" ::: "memory");
#endif
#else
        for (int i = tid; i < 2048; i += 128) l[i] = g[i];
#endif
    }
    __syncthreads();

    const int lane    = tid & 31;
    const int wave    = tid >> 5;
    const int rowBase = (blockIdx.x * 4 + wave) << 4;
    if (rowBase >= nRows) return;                 // wave-uniform exit
    const int l15   = lane & 15;
    const int half8 = (lane < 16) ? 0 : 8;
    const int mRow  = rowBase + l15;

    const float* arow = A + (size_t)mRow * 128;
    const float  s    = scale[mRow];

    // A fragments: elements 0..7 <- K = kt*32+half8+0..7 ; 8..15 <- +16
    v16bf afr[4];
#pragma unroll
    for (int kt = 0; kt < 4; ++kt) {
        const float* p = arow + kt * 32 + half8;
        float4 f0 = *(const float4*)(p);
        float4 f1 = *(const float4*)(p + 4);
        float4 f2 = *(const float4*)(p + 16);
        float4 f3 = *(const float4*)(p + 20);
        afr[kt][0]  = (__bf16)(f0.x * s);  afr[kt][1]  = (__bf16)(f0.y * s);
        afr[kt][2]  = (__bf16)(f0.z * s);  afr[kt][3]  = (__bf16)(f0.w * s);
        afr[kt][4]  = (__bf16)(f1.x * s);  afr[kt][5]  = (__bf16)(f1.y * s);
        afr[kt][6]  = (__bf16)(f1.z * s);  afr[kt][7]  = (__bf16)(f1.w * s);
        afr[kt][8]  = (__bf16)(f2.x * s);  afr[kt][9]  = (__bf16)(f2.y * s);
        afr[kt][10] = (__bf16)(f2.z * s);  afr[kt][11] = (__bf16)(f2.w * s);
        afr[kt][12] = (__bf16)(f3.x * s);  afr[kt][13] = (__bf16)(f3.y * s);
        afr[kt][14] = (__bf16)(f3.z * s);  afr[kt][15] = (__bf16)(f3.w * s);
    }

#pragma unroll 2
    for (int nt = 0; nt < 8; ++nt) {
        const int col = nt * 16 + l15;
        float* op = OUT + (size_t)(rowBase + half8) * 128 + col;
        const float bv = bias[col];

        v8f c;
        if constexpr (ACC) {
#pragma unroll
            for (int j = 0; j < 8; ++j) c[j] = op[(size_t)j * 128] + bv;
        } else {
#pragma unroll
            for (int j = 0; j < 8; ++j) c[j] = bv;
        }

        // preload all 4 B fragments (distinct regs) so ds loads overlap WMMA
        v16bf bfr[4];
#pragma unroll
        for (int kt = 0; kt < 4; ++kt)
            bfr[kt] = *(const v16bf*)(Wlds + ((size_t)(nt * 4 + kt) * 32 + lane) * 16);

#pragma unroll
        for (int kt = 0; kt < 4; ++kt)
            c = __builtin_amdgcn_wmma_f32_16x16x32_bf16(
                    false, afr[kt], false, bfr[kt], (short)0, c, false, false);

#pragma unroll
        for (int j = 0; j < 8; ++j) {
            float x = c[j];
            if constexpr (RELU) x = x > 0.0f ? x : 0.0f;
            op[(size_t)j * 128] = x;
        }
    }
}

// ---------------------------------------------------------------------------
// Host orchestration
// ---------------------------------------------------------------------------
extern "C" void kernel_launch(void* const* d_in, const int* in_sizes, int n_in,
                              void* d_out, int out_size, void* d_ws, size_t ws_size,
                              hipStream_t stream) {
    const float* x_user   = (const float*)d_in[0];
    const float* x_travel = (const float*)d_in[1];
    const float* x_visit  = (const float*)d_in[2];
    const int*   src_ut   = (const int*)d_in[3];
    const int*   dst_ut   = (const int*)d_in[4];
    const int*   src_tv   = (const int*)d_in[5];
    const int*   dst_tv   = (const int*)d_in[6];
    const float* We_user   = (const float*)d_in[7];
    const float* be_user   = (const float*)d_in[8];
    const float* We_travel = (const float*)d_in[9];
    const float* be_travel = (const float*)d_in[10];
    const float* We_visit  = (const float*)d_in[11];
    const float* be_visit  = (const float*)d_in[12];
    const float* Wl1 = (const float*)d_in[13];
    const float* bl1 = (const float*)d_in[14];
    const float* Wr1 = (const float*)d_in[15];
    const float* Wl2 = (const float*)d_in[16];
    const float* bl2 = (const float*)d_in[17];
    const float* Wr2 = (const float*)d_in[18];

    const int NU = 50000, NT = 100000, NV = 200000, H = 128;
    const int EUT = in_sizes[3];
    const int ETV = in_sizes[5];
    const int WSZ = H * H;  // 16384 elements per weight matrix

    // Workspace carve (base assumed 256B-aligned)
    char* ws = (char*)d_ws;
    size_t off = 0;
    auto carve = [&](size_t bytes) -> char* {
        char* p = ws + off;
        off += (bytes + 255) & ~(size_t)255;
        return p;
    };
    float* u0 = (float*)carve((size_t)NU * H * 4);
    float* t0 = (float*)carve((size_t)NT * H * 4);
    float* v0 = (float*)carve((size_t)NV * H * 4);
    float* u1 = (float*)carve((size_t)NU * H * 4);
    float* t1 = (float*)carve((size_t)NT * H * 4);
    float* v1 = (float*)carve((size_t)NV * H * 4);
    float* agg = (float*)carve((size_t)NV * H * 4);   // reused per direction
    float* rc_u   = (float*)carve((size_t)NU * 4);    // deg recip: t->u (dst=src_ut)
    float* rc_tu  = (float*)carve((size_t)NT * 4);    // u->t (dst=dst_ut)
    float* rc_tvt = (float*)carve((size_t)NT * 4);    // v->t (dst=src_tv)
    float* rc_v   = (float*)carve((size_t)NV * 4);    // t->v (dst=dst_tv)
    float* ones   = (float*)carve((size_t)NV * 4);    // unit row-scale
    float* zbias  = (float*)carve((size_t)H * 4);     // zero bias
    __bf16* pk = (__bf16*)carve((size_t)16 * WSZ * sizeof(__bf16));
    __bf16* pkWl1 = pk + 0 * 4 * WSZ;
    __bf16* pkWr1 = pk + 1 * 4 * WSZ;
    __bf16* pkWl2 = pk + 2 * 4 * WSZ;
    __bf16* pkWr2 = pk + 3 * 4 * WSZ;

    float* outU = (float*)d_out;
    float* outT = outU + (size_t)NU * H;
    float* outV = outT + (size_t)NT * H;

    auto fill = [&](float* p, float v, size_t n) {
        fillv_kernel<<<(unsigned)((n + 255) / 256), 256, 0, stream>>>(p, v, n);
    };
    auto gemm = [&](const float* A, const __bf16* Wp, const float* bias,
                    const float* sc, float* O, int rows, bool acc, bool relu) {
        const float* b = bias ? bias : zbias;
        const float* s = sc ? sc : ones;
        dim3 g((rows + 63) / 64), blk(128);
        if (acc) {
            if (relu) gemm_wmma_kernel<true, true><<<g, blk, 0, stream>>>(A, Wp, b, s, O, rows);
            else      gemm_wmma_kernel<true, false><<<g, blk, 0, stream>>>(A, Wp, b, s, O, rows);
        } else {
            if (relu) gemm_wmma_kernel<false, true><<<g, blk, 0, stream>>>(A, Wp, b, s, O, rows);
            else      gemm_wmma_kernel<false, false><<<g, blk, 0, stream>>>(A, Wp, b, s, O, rows);
        }
    };
    auto scat = [&](const float* X, const int* s, const int* d, float* a, int E) {
        long long thr = (long long)E * 32;
        scatter_add_kernel<<<(unsigned)((thr + 255) / 256), 256, 0, stream>>>(X, s, d, a, E);
    };

    // --- constants + pack all 16 HxH weights into WMMA B-fragment layout ---
    fill(ones, 1.0f, (size_t)NV);
    fill(zbias, 0.0f, (size_t)H);
    for (int m = 0; m < 4; ++m) {
        pack_w_kernel<<<4, 256, 0, stream>>>(Wl1 + (size_t)m * WSZ, pkWl1 + (size_t)m * WSZ);
        pack_w_kernel<<<4, 256, 0, stream>>>(Wr1 + (size_t)m * WSZ, pkWr1 + (size_t)m * WSZ);
        pack_w_kernel<<<4, 256, 0, stream>>>(Wl2 + (size_t)m * WSZ, pkWl2 + (size_t)m * WSZ);
        pack_w_kernel<<<4, 256, 0, stream>>>(Wr2 + (size_t)m * WSZ, pkWr2 + (size_t)m * WSZ);
    }

    // --- encoders ---
    encode_kernel<<<(NU + 31) / 32, 128, 0, stream>>>(x_user,   We_user,   be_user,   u0, NU, 17);
    encode_kernel<<<(NT + 31) / 32, 128, 0, stream>>>(x_travel, We_travel, be_travel, t0, NT, 21);
    encode_kernel<<<(NV + 31) / 32, 128, 0, stream>>>(x_visit,  We_visit,  be_visit,  v0, NV, 34);

    // --- degrees (shared by both layers) ---
    fill(rc_u, 0.0f, NU); fill(rc_tu, 0.0f, NT); fill(rc_tvt, 0.0f, NT); fill(rc_v, 0.0f, NV);
    count_kernel<<<(EUT + 255) / 256, 256, 0, stream>>>(src_ut, rc_u,   EUT);
    count_kernel<<<(EUT + 255) / 256, 256, 0, stream>>>(dst_ut, rc_tu,  EUT);
    count_kernel<<<(ETV + 255) / 256, 256, 0, stream>>>(src_tv, rc_tvt, ETV);
    count_kernel<<<(ETV + 255) / 256, 256, 0, stream>>>(dst_tv, rc_v,   ETV);
    recip_kernel<<<(NU + 255) / 256, 256, 0, stream>>>(rc_u,   NU);
    recip_kernel<<<(NT + 255) / 256, 256, 0, stream>>>(rc_tu,  NT);
    recip_kernel<<<(NT + 255) / 256, 256, 0, stream>>>(rc_tvt, NT);
    recip_kernel<<<(NV + 255) / 256, 256, 0, stream>>>(rc_v,   NV);

    // --- one hetero SAGE layer ---
    auto layer = [&](const float* ui, const float* ti, const float* vi,
                     float* uo, float* to, float* vo,
                     const __bf16* pWl, const float* bl, const __bf16* pWr, bool relu) {
        // out_u = mean_{t->u} @ Wl[1] + bl[1] + u @ Wr[1]
        fill(agg, 0.0f, (size_t)NU * H);
        scat(ti, dst_ut, src_ut, agg, EUT);
        gemm(agg, pWl + 1 * WSZ, bl + 1 * H, rc_u, uo, NU, false, false);
        gemm(ui,  pWr + 1 * WSZ, nullptr,  nullptr, uo, NU, true, relu);
        // out_t = mean_{u->t} @ Wl[0] + bl[0] + t @ Wr[0]
        //       + mean_{v->t} @ Wl[3] + bl[3] + t @ Wr[3]
        fill(agg, 0.0f, (size_t)NT * H);
        scat(ui, src_ut, dst_ut, agg, EUT);
        gemm(agg, pWl + 0 * WSZ, bl + 0 * H, rc_tu, to, NT, false, false);
        gemm(ti,  pWr + 0 * WSZ, nullptr,  nullptr, to, NT, true, false);
        fill(agg, 0.0f, (size_t)NT * H);
        scat(vi, dst_tv, src_tv, agg, ETV);
        gemm(agg, pWl + 3 * WSZ, bl + 3 * H, rc_tvt, to, NT, true, false);
        gemm(ti,  pWr + 3 * WSZ, nullptr,  nullptr, to, NT, true, relu);
        // out_v = mean_{t->v} @ Wl[2] + bl[2] + v @ Wr[2]
        fill(agg, 0.0f, (size_t)NV * H);
        scat(ti, src_tv, dst_tv, agg, ETV);
        gemm(agg, pWl + 2 * WSZ, bl + 2 * H, rc_v, vo, NV, false, false);
        gemm(vi,  pWr + 2 * WSZ, nullptr,  nullptr, vo, NV, true, relu);
    };

    layer(u0, t0, v0, u1, t1, v1, pkWl1, bl1, pkWr1, true);
    layer(u1, t1, v1, outU, outT, outV, pkWl2, bl2, pkWr2, false);
}
